// CrossAttention_9182640078960
// MI455X (gfx1250) — compile-verified, matrix operand order checked by reading, not run on hardware
//
#include <hip/hip_runtime.h>
#include <hip/hip_bf16.h>

// ---------------------------------------------------------------------------
// CDNA5 (gfx1250) cross-attention, bf16 WMMA path.
//   B=4, S1=S2=2048, IN=512, H=8, KD=VD=64
// ---------------------------------------------------------------------------

typedef __attribute__((ext_vector_type(16))) __bf16 v16bf;
typedef __attribute__((ext_vector_type(8)))  __bf16 v8bf;
typedef __attribute__((ext_vector_type(8)))  float  v8f;

#define S_LEN   2048
#define D_MODEL 512
#define NB      4
#define NH      8
#define HD      64

__device__ __forceinline__ v8f wmma_bf16(v16bf a, v16bf b, v8f c) {
  // D = A(16x32) * B(32x16) + C, f32 accumulate
  return __builtin_amdgcn_wmma_f32_16x16x32_bf16(
      /*neg_a=*/false, a, /*neg_b=*/false, b,
      /*c_mod=*/(short)0, c, /*reuse_a=*/false, /*reuse_b=*/false);
}

// Load a 16x32 (rows x contraction) fragment from a row-major [rows, ld] bf16
// matrix (global or LDS). Lane l holds row (row0 + (l&15)); contraction chunks
// at k = koff + (l>>4)*8 .. +7  and  k + 16 .. +23   (ISA 16-bit A/B layout).
__device__ __forceinline__ v16bf load_frag(const __bf16* __restrict__ base,
                                           int row0, int ld, int koff, int lane) {
  const int r  = row0 + (lane & 15);
  const int kb = koff + ((lane >> 4) << 3);
  const __bf16* p = base + (long)r * ld + kb;
  v8bf lo = *(const v8bf*)(p);
  v8bf hi = *(const v8bf*)(p + 16);
  v16bf f;
#pragma unroll
  for (int i = 0; i < 8; ++i) { f[i] = lo[i]; f[8 + i] = hi[i]; }
  return f;
}

// ---------------------------------------------------------------------------
// 1) f32 -> bf16 conversion
// ---------------------------------------------------------------------------
__global__ void cvt_f32_bf16(const float* __restrict__ src,
                             __bf16* __restrict__ dst, int n) {
  int i = blockIdx.x * blockDim.x + threadIdx.x;
  if (i < n) dst[i] = (__bf16)src[i];
}

// ---------------------------------------------------------------------------
// 2) Fused QKV projection GEMM.
//    A[M=8192, K=512] (x1 for Q, x2 for K/V), W[N=512, K=512] row-major.
//    Q,K written as [B,H,S,64]; V written transposed as [B,H,64,S].
//    Wave tile 32x64 (2x4 accumulators). Block = 4 waves -> 64x128 tile.
// ---------------------------------------------------------------------------
__global__ void qkv_gemm(const __bf16* __restrict__ x1b,
                         const __bf16* __restrict__ x2b,
                         const __bf16* __restrict__ Wqb,
                         const __bf16* __restrict__ Wkb,
                         const __bf16* __restrict__ Wvb,
                         __bf16* __restrict__ Qb,
                         __bf16* __restrict__ Kb,
                         __bf16* __restrict__ Vtb) {
  const int lane  = threadIdx.x & 31;
  const int wid   = threadIdx.x >> 5;   // 0..3
  const int which = blockIdx.z;         // 0=Q, 1=K, 2=V

  const __bf16* A = (which == 0) ? x1b : x2b;
  const __bf16* W = (which == 0) ? Wqb : (which == 1) ? Wkb : Wvb;
  __bf16*       D = (which == 0) ? Qb  : (which == 1) ? Kb  : Vtb;

  const int mOff = blockIdx.x * 64 + (wid & 1) * 32;
  const int nOff = blockIdx.y * 128 + (wid >> 1) * 64;

  v8f acc[2][4];
#pragma unroll
  for (int mt = 0; mt < 2; ++mt)
#pragma unroll
    for (int nt = 0; nt < 4; ++nt)
#pragma unroll
      for (int i = 0; i < 8; ++i) acc[mt][nt][i] = 0.0f;

  for (int k = 0; k < D_MODEL; k += 32) {
    v16bf af[2], bf_[4];
#pragma unroll
    for (int mt = 0; mt < 2; ++mt)
      af[mt] = load_frag(A, mOff + mt * 16, D_MODEL, k, lane);
#pragma unroll
    for (int nt = 0; nt < 4; ++nt)
      bf_[nt] = load_frag(W, nOff + nt * 16, D_MODEL, k, lane);
#pragma unroll
    for (int mt = 0; mt < 2; ++mt)
#pragma unroll
      for (int nt = 0; nt < 4; ++nt)
        acc[mt][nt] = wmma_bf16(af[mt], bf_[nt], acc[mt][nt]);
  }

  // C layout: lane<16 -> v[i] = C[m=i][n=lane]; lane>=16 -> C[m=8+i][n=lane-16]
  const int ncol = nOff + (lane & 15);
  const int mhi  = (lane >> 4) << 3;
#pragma unroll
  for (int mt = 0; mt < 2; ++mt) {
#pragma unroll
    for (int nt = 0; nt < 4; ++nt) {
      const int n = ncol + nt * 16;
      const int h = n >> 6, d = n & 63;
#pragma unroll
      for (int i = 0; i < 8; ++i) {
        const int m = mOff + mt * 16 + mhi + i;
        const int b = m >> 11, s = m & (S_LEN - 1);
        long addr;
        if (which == 2)  // V transposed: [B,H,64,S]
          addr = (((long)(b * NH + h) * HD + d) * S_LEN + s);
        else             // Q/K: [B,H,S,64]
          addr = (((long)(b * NH + h) * S_LEN + s) * HD + d);
        D[addr] = (__bf16)acc[mt][nt][i];
      }
    }
  }
}

// ---------------------------------------------------------------------------
// 3) Flash attention, LDS-staged.
//    Block = 8 waves, all on the SAME (b,h); each wave owns 32 q-rows.
//    Per 64-key step the block cooperatively stages K(64x64) and V^T(64x64)
//    tiles into LDS (16 KB of the 320 KB WGP LDS), then each wave does
//    32 WMMAs against them (2 x 32-key halves):
//      S^T(32x16 x2 q-tiles) = K_tile * Q^T          (8 WMMA / half)
//      online softmax (local + shfl_xor(16) only)
//      O^T(64x16 x2 q-tiles) += V^T * P              (8 WMMA / half)
//    P lands in exactly the B-operand layout -> no lane shuffles of P.
//    Next tile's global loads are issued right after the barrier so they
//    overlap the compute of the current tile.
// ---------------------------------------------------------------------------
__global__ void attn_flash(const __bf16* __restrict__ Qb,
                           const __bf16* __restrict__ Kb,
                           const __bf16* __restrict__ Vtb,
                           __bf16* __restrict__ AOb) {
  const int lane = threadIdx.x & 31;
  const int wid  = threadIdx.x >> 5;              // 0..7
  const int tid  = threadIdx.x;                   // 0..255
  const int tile = blockIdx.x * 8 + wid;          // 0..2047 (32-q tiles)
  const int bh   = tile >> 6;                     // 64 tiles per (b,h)
  const int q0   = (tile & 63) << 5;              // 0..2016

  const __bf16* Qp = Qb  + (long)bh * S_LEN * HD;
  const __bf16* Kp = Kb  + (long)bh * S_LEN * HD;   // uniform across block
  const __bf16* Vp = Vtb + (long)bh * HD * S_LEN;   // uniform across block

  __shared__ __bf16 Ks[64 * 64];   // [key][d]
  __shared__ __bf16 Vs[64 * 64];   // [d][key]

  // Q fragments: B-operand of S^T (lane = q column, contraction = d)
  v16bf qf[2][2];
#pragma unroll
  for (int t = 0; t < 2; ++t) {
    qf[t][0] = load_frag(Qp, q0 + t * 16, HD, 0,  lane);
    qf[t][1] = load_frag(Qp, q0 + t * 16, HD, 32, lane);
  }

  v8f acc[2][4];
#pragma unroll
  for (int t = 0; t < 2; ++t)
#pragma unroll
    for (int dt = 0; dt < 4; ++dt)
#pragma unroll
      for (int i = 0; i < 8; ++i) acc[t][dt][i] = 0.0f;

  float mrun[2] = {-INFINITY, -INFINITY};
  float lsum[2] = {0.0f, 0.0f};
  const float scl = 0.125f;  // 1/sqrt(64)

  // Cooperative tile staging: K tile is 4096 contiguous bf16 (rows j..j+63);
  // V^T tile is 64 rows (d) x 64 keys with row stride S_LEN.
  v8bf kr0, kr1, vr0, vr1;
  const int vrow = tid >> 2;            // 0..63  (d)
  const int vcol = (tid & 3) * 16;      // 0,16,32,48 (key)
  auto load_tile_regs = [&](int j) {
    kr0 = *(const v8bf*)(Kp + (long)j * HD + tid * 8);
    kr1 = *(const v8bf*)(Kp + (long)j * HD + 2048 + tid * 8);
    const __bf16* vp = Vp + (long)vrow * S_LEN + j + vcol;
    vr0 = *(const v8bf*)(vp);
    vr1 = *(const v8bf*)(vp + 8);
  };
  load_tile_regs(0);

  for (int j = 0; j < S_LEN; j += 64) {
    __syncthreads();                       // everyone done with previous tile
    *(v8bf*)(Ks + tid * 8)        = kr0;
    *(v8bf*)(Ks + 2048 + tid * 8) = kr1;
    {
      __bf16* vd = Vs + vrow * 64 + vcol;
      *(v8bf*)(vd)     = vr0;
      *(v8bf*)(vd + 8) = vr1;
    }
    __syncthreads();                       // tile visible to all waves
    if (j + 64 < S_LEN) load_tile_regs(j + 64);   // overlap next tile's loads

#pragma unroll
    for (int half = 0; half < 2; ++half) {
      const int jl = half * 32;            // local key offset in LDS tile

      // K fragments from LDS (shared by both q-tiles)
      v16bf k0a = load_frag(Ks, jl,      64, 0,  lane);
      v16bf k0b = load_frag(Ks, jl,      64, 32, lane);
      v16bf k1a = load_frag(Ks, jl + 16, 64, 0,  lane);
      v16bf k1b = load_frag(Ks, jl + 16, 64, 32, lane);

      v16bf pb[2];
#pragma unroll
      for (int t = 0; t < 2; ++t) {
        v8f st0, st1;
#pragma unroll
        for (int i = 0; i < 8; ++i) { st0[i] = 0.0f; st1[i] = 0.0f; }
        st0 = wmma_bf16(k0a, qf[t][0], st0);
        st0 = wmma_bf16(k0b, qf[t][1], st0);
        st1 = wmma_bf16(k1a, qf[t][0], st1);
        st1 = wmma_bf16(k1b, qf[t][1], st1);

        // online softmax (column q = lane pair {l, l+16})
        float sc0[8], sc1[8];
        float mloc = -INFINITY;
#pragma unroll
        for (int i = 0; i < 8; ++i) {
          sc0[i] = st0[i] * scl; mloc = fmaxf(mloc, sc0[i]);
          sc1[i] = st1[i] * scl; mloc = fmaxf(mloc, sc1[i]);
        }
        mloc = fmaxf(mloc, __shfl_xor(mloc, 16, 32));
        const float mnew = fmaxf(mrun[t], mloc);
        const float corr = __expf(mrun[t] - mnew);

        float ps = 0.0f;
#pragma unroll
        for (int i = 0; i < 8; ++i) {
          float p = __expf(sc0[i] - mnew); ps += p; pb[t][i]     = (__bf16)p;
        }
#pragma unroll
        for (int i = 0; i < 8; ++i) {
          float p = __expf(sc1[i] - mnew); ps += p; pb[t][8 + i] = (__bf16)p;
        }
        ps += __shfl_xor(ps, 16, 32);
        lsum[t] = lsum[t] * corr + ps;
        mrun[t] = mnew;

#pragma unroll
        for (int dt = 0; dt < 4; ++dt)
#pragma unroll
          for (int i = 0; i < 8; ++i) acc[t][dt][i] *= corr;
      }

      // O^T += V^T * P  (V fragment shared by both q-tiles)
#pragma unroll
      for (int dt = 0; dt < 4; ++dt) {
        v16bf vf = load_frag(Vs, dt * 16, 64, jl, lane);
#pragma unroll
        for (int t = 0; t < 2; ++t)
          acc[t][dt] = wmma_bf16(vf, pb[t], acc[t][dt]);
      }
    }
  }

  // ----- epilogue: O = acc / lsum, write [B, S1, H*64] bf16 -----
  const int b = bh >> 3, h = bh & 7;
  const int dhi = (lane >> 4) << 3;
#pragma unroll
  for (int t = 0; t < 2; ++t) {
    const float inv = 1.0f / lsum[t];
    const int qcol = q0 + t * 16 + (lane & 15);
#pragma unroll
    for (int dt = 0; dt < 4; ++dt) {
#pragma unroll
      for (int i = 0; i < 8; ++i) {
        const int d = dt * 16 + dhi + i;
        AOb[((long)(b * S_LEN) + qcol) * D_MODEL + h * HD + d] =
            (__bf16)(acc[t][dt][i] * inv);
      }
    }
  }
}

// ---------------------------------------------------------------------------
// 4) Output projection: out[8192,512] = AO[8192,512] * Wo^T + bo   (f32 out)
// ---------------------------------------------------------------------------
__global__ void out_gemm(const __bf16* __restrict__ AOb,
                         const __bf16* __restrict__ Wob,
                         const float* __restrict__ bo,
                         float* __restrict__ out) {
  const int lane = threadIdx.x & 31;
  const int wid  = threadIdx.x >> 5;

  const int mOff = blockIdx.x * 64 + (wid & 1) * 32;
  const int nOff = blockIdx.y * 128 + (wid >> 1) * 64;

  v8f acc[2][4];
#pragma unroll
  for (int mt = 0; mt < 2; ++mt)
#pragma unroll
    for (int nt = 0; nt < 4; ++nt)
#pragma unroll
      for (int i = 0; i < 8; ++i) acc[mt][nt][i] = 0.0f;

  for (int k = 0; k < D_MODEL; k += 32) {
    v16bf af[2], bf_[4];
#pragma unroll
    for (int mt = 0; mt < 2; ++mt)
      af[mt] = load_frag(AOb, mOff + mt * 16, D_MODEL, k, lane);
#pragma unroll
    for (int nt = 0; nt < 4; ++nt)
      bf_[nt] = load_frag(Wob, nOff + nt * 16, D_MODEL, k, lane);
#pragma unroll
    for (int mt = 0; mt < 2; ++mt)
#pragma unroll
      for (int nt = 0; nt < 4; ++nt)
        acc[mt][nt] = wmma_bf16(af[mt], bf_[nt], acc[mt][nt]);
  }

  const int ncol = nOff + (lane & 15);
  const int mhi  = (lane >> 4) << 3;
#pragma unroll
  for (int mt = 0; mt < 2; ++mt) {
#pragma unroll
    for (int nt = 0; nt < 4; ++nt) {
      const int n = ncol + nt * 16;
      const float bias = bo[n];
#pragma unroll
      for (int i = 0; i < 8; ++i) {
        const int m = mOff + mt * 16 + mhi + i;
        out[(long)m * D_MODEL + n] = acc[mt][nt][i] + bias;
      }
    }
  }
}

// ---------------------------------------------------------------------------
// Host launcher
// ---------------------------------------------------------------------------
extern "C" void kernel_launch(void* const* d_in, const int* in_sizes, int n_in,
                              void* d_out, int out_size, void* d_ws, size_t ws_size,
                              hipStream_t stream) {
  const float* x1 = (const float*)d_in[0];
  const float* x2 = (const float*)d_in[1];
  const float* Wq = (const float*)d_in[2];
  const float* Wk = (const float*)d_in[3];
  const float* Wv = (const float*)d_in[4];
  const float* Wo = (const float*)d_in[5];
  const float* bo = (const float*)d_in[6];
  float* out = (float*)d_out;

  const long N_ACT = (long)NB * S_LEN * D_MODEL;   // 4,194,304
  const long N_W   = (long)D_MODEL * D_MODEL;      //   262,144

  __bf16* ws  = (__bf16*)d_ws;
  __bf16* x1b = ws;                 // [8192, 512]
  __bf16* x2b = x1b + N_ACT;        // [8192, 512]
  __bf16* Wqb = x2b + N_ACT;
  __bf16* Wkb = Wqb + N_W;
  __bf16* Wvb = Wkb + N_W;
  __bf16* Wob = Wvb + N_W;
  __bf16* Qb  = Wob + N_W;          // [B,H,S,64]
  __bf16* Kb  = Qb  + N_ACT;        // [B,H,S,64]
  __bf16* Vtb = Kb  + N_ACT;        // [B,H,64,S]
  __bf16* AOb = Vtb + N_ACT;        // [8192, 512]

  const int CT = 256;
  cvt_f32_bf16<<<(int)((N_ACT + CT - 1) / CT), CT, 0, stream>>>(x1, x1b, (int)N_ACT);
  cvt_f32_bf16<<<(int)((N_ACT + CT - 1) / CT), CT, 0, stream>>>(x2, x2b, (int)N_ACT);
  cvt_f32_bf16<<<(int)((N_W + CT - 1) / CT),   CT, 0, stream>>>(Wq, Wqb, (int)N_W);
  cvt_f32_bf16<<<(int)((N_W + CT - 1) / CT),   CT, 0, stream>>>(Wk, Wkb, (int)N_W);
  cvt_f32_bf16<<<(int)((N_W + CT - 1) / CT),   CT, 0, stream>>>(Wv, Wvb, (int)N_W);
  cvt_f32_bf16<<<(int)((N_W + CT - 1) / CT),   CT, 0, stream>>>(Wo, Wob, (int)N_W);

  // QKV projections: M=8192 (blocks of 64), N=512 (blocks of 128), z selects Q/K/V
  qkv_gemm<<<dim3(8192 / 64, 512 / 128, 3), 128, 0, stream>>>(
      x1b, x2b, Wqb, Wkb, Wvb, Qb, Kb, Vtb);

  // Attention: 2048 q-tiles of 32 rows, 8 waves (one (b,h)) per block
  attn_flash<<<2048 / 8, 256, 0, stream>>>(Qb, Kb, Vtb, AOb);

  // Output projection + bias
  out_gemm<<<dim3(8192 / 64, 512 / 128), 128, 0, stream>>>(AOb, Wob, bo, out);
}
